// TextCrossAttention_32195074851203
// MI455X (gfx1250) — matrix operand order repaired and checked by reading, not compiled, not run on hardware
//
#include <hip/hip_runtime.h>
#include <hip/hip_bf16.h>

typedef __attribute__((ext_vector_type(16))) __bf16 v16bf;
typedef __attribute__((ext_vector_type(8)))  float  v8f;

#define D384 384
#define NHEADS 8
#define HDIM 48
#define KNN 10
#define SEQ 14
#define NBATCH 64
#define NDOCS 8192
#define ROWS 8960   // 64*10*14 tokens

// ---------------------------------------------------------------------------
// Generic GEMM: C[M,N] = act( A[M,K] @ W[N,K]^T + bias[N] + res[M,N] )
// f32 in memory, bf16 in registers, f32 WMMA accumulate.
// Each wave computes a 32x32 output tile: 2 A-frags x 2 B-frags -> 4 accs,
// 4 x v_wmma_f32_16x16x32_bf16 per K-step. 4 waves/block -> 32x128 block tile.
// Out-of-range A rows are CLAMPED (not branched): they only feed C rows that
// the store guard discards, so the K-loop runs with full EXEC, branch-free.
// act: 0 = none, 1 = exact GELU.
// ---------------------------------------------------------------------------
__global__ void gemm_bf16_wmma(const float* __restrict__ A, int lda, long long sA,
                               const float* __restrict__ W, int ldw, long long sW,
                               const float* __restrict__ bias,
                               const float* __restrict__ res, long long sRes,
                               float* __restrict__ C, int ldc, long long sC,
                               int M, int N, int K, int act) {
    const int lane = threadIdx.x & 31;
    const int wave = threadIdx.x >> 5;
    const int b    = blockIdx.z;

    A += (long long)b * sA;
    W += (long long)b * sW;
    C += (long long)b * sC;
    if (res) res += (long long)b * sRes;

    const int tm = blockIdx.y * 32;
    const int tn = blockIdx.x * 128 + wave * 32;
    if (tn >= N) return;                       // wave-uniform

    // A fragment rows (clamped to valid memory; results guarded at store)
    int r0 = tm + (lane & 15);       if (r0 >= M) r0 = M - 1;
    int r1 = tm + 16 + (lane & 15);  if (r1 >= M) r1 = M - 1;
    const int n0 = tn + (lane & 15);
    const int n1 = n0 + 16;

    const int kbA = (lane < 16) ? 0 : 16;      // contiguous k layout (matches B)
    const int kbB = (lane < 16) ? 0 : 16;      // contiguous k layout (matches A)

    const float* a0 = A + (long long)r0 * lda + kbA;
    const float* a1 = A + (long long)r1 * lda + kbA;
    const float* b0 = W + (long long)n0 * ldw + kbB;
    const float* b1 = W + (long long)n1 * ldw + kbB;

    v8f acc00 = {}, acc01 = {}, acc10 = {}, acc11 = {};

    for (int k0 = 0; k0 < K; k0 += 32) {
        __builtin_prefetch(b0 + k0 + 128, 0, 1);   // global_prefetch_b8
        __builtin_prefetch(b1 + k0 + 128, 0, 1);
        __builtin_prefetch(a0 + k0 + 128, 0, 1);

        v16bf af0, af1, bf0, bf1;
        {
            const float* p = a0 + k0;
            #pragma unroll
            for (int i = 0; i < 8; ++i) af0[i]     = (__bf16)p[i];        // K=kbA+0..7
            #pragma unroll
            for (int i = 0; i < 8; ++i) af0[8 + i] = (__bf16)p[8 + i];    // K=kbA+8..15
        }
        {
            const float* p = a1 + k0;
            #pragma unroll
            for (int i = 0; i < 8; ++i) af1[i]     = (__bf16)p[i];
            #pragma unroll
            for (int i = 0; i < 8; ++i) af1[8 + i] = (__bf16)p[8 + i];
        }
        {
            const float* p = b0 + k0;
            #pragma unroll
            for (int i = 0; i < 16; ++i) bf0[i] = (__bf16)p[i];           // K=kbB+0..15
        }
        {
            const float* p = b1 + k0;
            #pragma unroll
            for (int i = 0; i < 16; ++i) bf1[i] = (__bf16)p[i];
        }

        acc00 = __builtin_amdgcn_wmma_f32_16x16x32_bf16(false, af0, false, bf0, (short)0, acc00, false, false);
        acc01 = __builtin_amdgcn_wmma_f32_16x16x32_bf16(false, af0, false, bf1, (short)0, acc01, false, false);
        acc10 = __builtin_amdgcn_wmma_f32_16x16x32_bf16(false, af1, false, bf0, (short)0, acc10, false, false);
        acc11 = __builtin_amdgcn_wmma_f32_16x16x32_bf16(false, af1, false, bf1, (short)0, acc11, false, false);
    }

    const float bias0 = bias ? bias[n0] : 0.0f;
    const float bias1 = bias ? bias[n1] : 0.0f;
    const int   mb0   = tm + ((lane < 16) ? 0 : 8);       // rows for acc0x
    const int   mb1   = mb0 + 16;                          // rows for acc1x

    #pragma unroll
    for (int r = 0; r < 8; ++r) {
        const int m0 = mb0 + r;
        if (m0 < M) {
            float v0 = acc00[r] + bias0;
            float v1 = acc01[r] + bias1;
            if (res) {
                v0 += res[(long long)m0 * ldc + n0];
                v1 += res[(long long)m0 * ldc + n1];
            }
            if (act == 1) {
                v0 = 0.5f * v0 * (1.0f + erff(v0 * 0.70710678118654752f));
                v1 = 0.5f * v1 * (1.0f + erff(v1 * 0.70710678118654752f));
            }
            C[(long long)m0 * ldc + n0] = v0;
            C[(long long)m0 * ldc + n1] = v1;
        }
        const int m1 = mb1 + r;
        if (m1 < M) {
            float v0 = acc10[r] + bias0;
            float v1 = acc11[r] + bias1;
            if (res) {
                v0 += res[(long long)m1 * ldc + n0];
                v1 += res[(long long)m1 * ldc + n1];
            }
            if (act == 1) {
                v0 = 0.5f * v0 * (1.0f + erff(v0 * 0.70710678118654752f));
                v1 = 0.5f * v1 * (1.0f + erff(v1 * 0.70710678118654752f));
            }
            C[(long long)m1 * ldc + n0] = v0;
            C[(long long)m1 * ldc + n1] = v1;
        }
    }
}

// ---------------------------------------------------------------------------
// top-10 of each 8192-wide row (896 rows); iterative masked argmax, descending
// values, lower index wins ties (matches jax.lax.top_k).
// ---------------------------------------------------------------------------
__global__ void topk10(const float* __restrict__ sim, int* __restrict__ idxOut) {
    const int row = blockIdx.x;                 // b*14 + h
    const float* s = sim + (long long)row * NDOCS;
    __shared__ float sval[256];
    __shared__ int   sidx[256];
    __shared__ int   chosen[KNN];
    const int t = threadIdx.x;

    for (int p = 0; p < KNN; ++p) {
        int ch[KNN];
        for (int c = 0; c < p; ++c) ch[c] = chosen[c];
        float best = -3.402823466e38f;
        int   bi   = 0x7fffffff;
        for (int i = t; i < NDOCS; i += 256) {
            float v = s[i];
            bool excl = false;
            for (int c = 0; c < p; ++c) excl |= (ch[c] == i);
            if (!excl && (v > best || (v == best && i < bi))) { best = v; bi = i; }
        }
        sval[t] = best; sidx[t] = bi;
        __syncthreads();
        for (int st = 128; st > 0; st >>= 1) {
            if (t < st) {
                float v2 = sval[t + st]; int i2 = sidx[t + st];
                if (v2 > sval[t] || (v2 == sval[t] && i2 < sidx[t])) {
                    sval[t] = v2; sidx[t] = i2;
                }
            }
            __syncthreads();
        }
        if (t == 0) { chosen[p] = sidx[0]; idxOut[row * KNN + p] = sidx[0]; }
        __syncthreads();
    }
}

// ---------------------------------------------------------------------------
// Build x0 = broadcast(qt) and mem = gathered nd rows.
// block (t in 0..13, j in 0..9, b in 0..63)
// ---------------------------------------------------------------------------
__global__ void gather_build(const float* __restrict__ nd, const float* __restrict__ qt,
                             const int* __restrict__ idx,
                             float* __restrict__ X, float* __restrict__ MEMB) {
    const int t = blockIdx.x, j = blockIdx.y, b = blockIdx.z;
    const int n = idx[(b * SEQ + t) * KNN + j];
    const float* src  = nd + ((long long)b * NDOCS + n) * D384;
    const float* qsrc = qt + ((long long)b * SEQ + t) * D384;
    const long long orow = (((long long)b * KNN + j) * SEQ + t) * D384;
    for (int c = threadIdx.x; c < D384; c += blockDim.x) {
        MEMB[orow + c] = src[c];
        X[orow + c]    = qsrc[c];
    }
}

// ---------------------------------------------------------------------------
// LayerNorm over last dim (384). One row per wave; 8 waves / block.
// Optional second output for the mem2 snapshot.
// ---------------------------------------------------------------------------
__global__ void layernorm384(const float* __restrict__ Xin, const float* __restrict__ w,
                             const float* __restrict__ bb, float* __restrict__ Y,
                             float* __restrict__ Y2, int rows) {
    const int wave = threadIdx.x >> 5;
    const int lane = threadIdx.x & 31;
    const long long row = (long long)blockIdx.x * (blockDim.x >> 5) + wave;
    if (row >= rows) return;
    const float* x = Xin + row * D384;

    float v[12];
    float s = 0.0f;
    #pragma unroll
    for (int i = 0; i < 12; ++i) { v[i] = x[lane + i * 32]; s += v[i]; }
    #pragma unroll
    for (int m = 16; m > 0; m >>= 1) s += __shfl_xor(s, m, 32);
    const float mu = s * (1.0f / D384);

    float qs = 0.0f;
    #pragma unroll
    for (int i = 0; i < 12; ++i) { float d = v[i] - mu; qs += d * d; }
    #pragma unroll
    for (int m = 16; m > 0; m >>= 1) qs += __shfl_xor(qs, m, 32);
    const float inv = rsqrtf(qs * (1.0f / D384) + 1e-5f);

    #pragma unroll
    for (int i = 0; i < 12; ++i) {
        const int c = lane + i * 32;
        const float o = (v[i] - mu) * inv * w[c] + bb[c];
        Y[row * D384 + c] = o;
        if (Y2) Y2[row * D384 + c] = o;
    }
}

// ---------------------------------------------------------------------------
// MHA core: seq=14, head_dim=48. One wave per (sequence, head); K,V in LDS.
// ---------------------------------------------------------------------------
__global__ void mha_attn(const float* __restrict__ Q, const float* __restrict__ K,
                         const float* __restrict__ V, float* __restrict__ O) {
    const int s = blockIdx.x;   // 0..639
    const int h = blockIdx.y;   // 0..7
    __shared__ float sk[SEQ * HDIM];
    __shared__ float sv[SEQ * HDIM];
    const int lane = threadIdx.x;
    const long long base = (long long)s * SEQ * D384 + h * HDIM;

    for (int i = lane; i < SEQ * HDIM; i += 32) {
        const int t = i / HDIM, d = i % HDIM;
        sk[i] = K[base + (long long)t * D384 + d];
        sv[i] = V[base + (long long)t * D384 + d];
    }
    __syncthreads();

    if (lane < SEQ) {
        float q[HDIM];
        #pragma unroll
        for (int d = 0; d < HDIM; ++d) q[d] = Q[base + (long long)lane * D384 + d];

        float sc[SEQ];
        float mx = -3.402823466e38f;
        #pragma unroll
        for (int t = 0; t < SEQ; ++t) {
            float a = 0.0f;
            #pragma unroll
            for (int d = 0; d < HDIM; ++d) a += q[d] * sk[t * HDIM + d];
            a *= 0.14433756729740643f;   // 1/sqrt(48)
            sc[t] = a;
            mx = fmaxf(mx, a);
        }
        float sum = 0.0f;
        #pragma unroll
        for (int t = 0; t < SEQ; ++t) { sc[t] = __expf(sc[t] - mx); sum += sc[t]; }
        const float r = 1.0f / sum;

        float o[HDIM];
        #pragma unroll
        for (int d = 0; d < HDIM; ++d) o[d] = 0.0f;
        #pragma unroll
        for (int t = 0; t < SEQ; ++t) {
            const float a = sc[t] * r;
            #pragma unroll
            for (int d = 0; d < HDIM; ++d) o[d] += a * sv[t * HDIM + d];
        }
        #pragma unroll
        for (int d = 0; d < HDIM; ++d) O[base + (long long)lane * D384 + d] = o[d];
    }
}

// ---------------------------------------------------------------------------
// Host orchestration
// ---------------------------------------------------------------------------
static inline void run_gemm(const float* A, int lda, long long sA,
                            const float* W, int ldw, long long sW,
                            const float* bias, const float* res, long long sRes,
                            float* C, int ldc, long long sC,
                            int M, int N, int K, int act, int batch, hipStream_t st) {
    dim3 g((N + 127) / 128, (M + 31) / 32, batch);
    gemm_bf16_wmma<<<g, 128, 0, st>>>(A, lda, sA, W, ldw, sW, bias, res, sRes,
                                      C, ldc, sC, M, N, K, act);
}

extern "C" void kernel_launch(void* const* d_in, const int* in_sizes, int n_in,
                              void* d_out, int out_size, void* d_ws, size_t ws_size,
                              hipStream_t stream) {
    const float* qt_emb = (const float*)d_in[0];   // (64,1,14,384) == (64,14,384)
    const float* nd_emb = (const float*)d_in[1];   // (64,8192,1,384) == (64,8192,384)
    const float* sa_w  = (const float*)d_in[2];
    const float* sa_b  = (const float*)d_in[3];
    const float* sa_ow = (const float*)d_in[4];
    const float* sa_ob = (const float*)d_in[5];
    const float* ca_w  = (const float*)d_in[6];
    const float* ca_b  = (const float*)d_in[7];
    const float* ca_ow = (const float*)d_in[8];
    const float* ca_ob = (const float*)d_in[9];
    const float* ln1w = (const float*)d_in[10];
    const float* ln1b = (const float*)d_in[11];
    const float* ln2w = (const float*)d_in[12];
    const float* ln2b = (const float*)d_in[13];
    const float* ln3w = (const float*)d_in[14];
    const float* ln3b = (const float*)d_in[15];
    const float* f1w = (const float*)d_in[16];
    const float* f1b = (const float*)d_in[17];
    const float* f2w = (const float*)d_in[18];
    const float* f2b = (const float*)d_in[19];
    const float* fnw = (const float*)d_in[20];
    const float* fnb = (const float*)d_in[21];

    float* ws = (float*)d_ws;
    // workspace layout (float offsets); sim region reused for Q/K after gather
    const long long SIM_O  = 0;                      // 64*14*8192 = 7,340,032
    const long long Q_O    = 0;                      // 3,440,640 (reuse of sim)
    const long long K_O    = 3440640;                // 3,440,640 (reuse of sim)
    const long long X_O    = 7340032;
    const long long H_O    = 10780672;
    const long long MEM_O  = 14221312;
    const long long M2_O   = 17661952;
    const long long V_O    = 21102592;
    const long long ATT_O  = 24543232;
    const long long FF_O   = 27983872;               // 8960*1536 = 13,762,560
    const long long IDX_O  = 41746432;               // 8960 ints

    float* SIM  = ws + SIM_O;
    float* Qb   = ws + Q_O;
    float* Kb   = ws + K_O;
    float* X    = ws + X_O;
    float* H    = ws + H_O;
    float* MEMB = ws + MEM_O;
    float* M2   = ws + M2_O;
    float* Vb   = ws + V_O;
    float* ATT  = ws + ATT_O;
    float* FF   = ws + FF_O;
    int*   IDX  = (int*)(ws + IDX_O);

    // 1) sim[b,h,n] = qt[b,h,:] . nd[b,n,:]   (batched GEMM, streams 806 MB)
    run_gemm(qt_emb, D384, (long long)SEQ * D384,
             nd_emb, D384, (long long)NDOCS * D384,
             nullptr, nullptr, 0,
             SIM, NDOCS, (long long)SEQ * NDOCS,
             SEQ, NDOCS, D384, 0, NBATCH, stream);

    // 2) top-10 per (b,h)
    topk10<<<NBATCH * SEQ, 256, 0, stream>>>(SIM, IDX);

    // 3) gather mem, broadcast x0
    gather_build<<<dim3(SEQ, KNN, NBATCH), 128, 0, stream>>>(nd_emb, qt_emb, IDX, X, MEMB);

    const int LNBLK = (ROWS + 7) / 8;   // 8 rows (waves) per 256-thread block

    for (int i = 0; i < 6; ++i) {
        const float* mem = (i < 3) ? MEMB : M2;
        const float* wqkv = sa_w + (long long)i * 1152 * D384;
        const float* bqkv = sa_b + (long long)i * 1152;
        const float* wc   = ca_w + (long long)i * 1152 * D384;
        const float* bc   = ca_b + (long long)i * 1152;

        // ---- self attention ----
        layernorm384<<<LNBLK, 256, 0, stream>>>(X, ln1w + i * D384, ln1b + i * D384, H, nullptr, ROWS);
        run_gemm(H, D384, 0, wqkv,                         D384, 0, bqkv,        nullptr, 0, Qb, D384, 0, ROWS, D384, D384, 0, 1, stream);
        run_gemm(H, D384, 0, wqkv + (long long)384 * D384, D384, 0, bqkv + 384,  nullptr, 0, Kb, D384, 0, ROWS, D384, D384, 0, 1, stream);
        run_gemm(H, D384, 0, wqkv + (long long)768 * D384, D384, 0, bqkv + 768,  nullptr, 0, Vb, D384, 0, ROWS, D384, D384, 0, 1, stream);
        mha_attn<<<dim3(NBATCH * KNN, NHEADS), 32, 0, stream>>>(Qb, Kb, Vb, ATT);
        run_gemm(ATT, D384, 0, sa_ow + (long long)i * D384 * D384, D384, 0,
                 sa_ob + (long long)i * D384, X, 0, X, D384, 0, ROWS, D384, D384, 0, 1, stream);

        // ---- cross attention ----
        layernorm384<<<LNBLK, 256, 0, stream>>>(X, ln2w + i * D384, ln2b + i * D384, H, nullptr, ROWS);
        run_gemm(H,   D384, 0, wc,                         D384, 0, bc,        nullptr, 0, Qb, D384, 0, ROWS, D384, D384, 0, 1, stream);
        run_gemm(mem, D384, 0, wc + (long long)384 * D384, D384, 0, bc + 384,  nullptr, 0, Kb, D384, 0, ROWS, D384, D384, 0, 1, stream);
        run_gemm(mem, D384, 0, wc + (long long)768 * D384, D384, 0, bc + 768,  nullptr, 0, Vb, D384, 0, ROWS, D384, D384, 0, 1, stream);
        mha_attn<<<dim3(NBATCH * KNN, NHEADS), 32, 0, stream>>>(Qb, Kb, Vb, ATT);
        run_gemm(ATT, D384, 0, ca_ow + (long long)i * D384 * D384, D384, 0,
                 ca_ob + (long long)i * D384, X, 0, X, D384, 0, ROWS, D384, D384, 0, 1, stream);

        // ---- FFN ----
        layernorm384<<<LNBLK, 256, 0, stream>>>(X, ln3w + i * D384, ln3b + i * D384, H, nullptr, ROWS);
        run_gemm(H,  D384, 0, f1w + (long long)i * 1536 * D384, D384, 0,
                 f1b + (long long)i * 1536, nullptr, 0, FF, 1536, 0, ROWS, 1536, D384, 1, 1, stream);
        run_gemm(FF, 1536, 0, f2w + (long long)i * D384 * 1536, 1536, 0,
                 f2b + (long long)i * D384, X, 0, X, D384, 0, ROWS, D384, 1536, 0, 1, stream);

        // snapshot mem2 after block 2
        if (i == 2) {
            layernorm384<<<LNBLK, 256, 0, stream>>>(X, fnw, fnb, X, M2, ROWS);
        }
    }

    // final LN -> d_out
    layernorm384<<<LNBLK, 256, 0, stream>>>(X, fnw + D384, fnb + D384, (float*)d_out, nullptr, ROWS);
}
